// DecoderLayer_46033459478661
// MI455X (gfx1250) — matrix operand order repaired and checked by reading, not compiled, not run on hardware
//
#include <hip/hip_runtime.h>
#include <hip/hip_bf16.h>

// ---------------------------------------------------------------------------
// Types for CDNA5 WMMA (wave32)
// ---------------------------------------------------------------------------
typedef __attribute__((ext_vector_type(16))) __bf16 v16bf;
typedef __attribute__((ext_vector_type(8)))  __bf16 bf16x8;
typedef __attribute__((ext_vector_type(8)))  float  v8f;

#define D_MODEL 256
#define NHEAD   8
#define HEAD_DIM 32
#define LSEQ    4800
#define NBATCH  8
#define MROWS   (NBATCH * LSEQ)          // 38400
#define KV_SPLIT 10

// ---------------------------------------------------------------------------
// Weight pre-swizzle: fp32 row-major W[K][N] -> bf16 WMMA-B layout
// dst flat f: j=f&15, lane=(f>>4)&31, rest=f>>9, kc=rest%(K/32), nt=rest/(K/32)
// element (k,n): n = nt*16 + (lane&15); k = kc*32 + ((lane>>4)<<4) + j
// ---------------------------------------------------------------------------
__global__ void prep_weight(const float* __restrict__ W, __bf16* __restrict__ Wsw,
                            int K, int N) {
    int f = blockIdx.x * blockDim.x + threadIdx.x;
    if (f >= K * N) return;
    int j    = f & 15;
    int lane = (f >> 4) & 31;
    int rest = f >> 9;
    int kcN  = K >> 5;
    int kc   = rest % kcN;
    int nt   = rest / kcN;
    int n = nt * 16 + (lane & 15);
    int k = kc * 32 + ((lane >> 4) << 4) + j;
    Wsw[f] = (__bf16)W[(size_t)k * N + n];
}

// ---------------------------------------------------------------------------
// Fused LayerNorm: one wave (32 lanes) per 256-wide row.
// Writes bf16 LN(x) and/or bf16 LN(x)+pos.
// ---------------------------------------------------------------------------
__global__ __launch_bounds__(256) void ln_fused(
    const float* __restrict__ x, const float* __restrict__ g, const float* __restrict__ b,
    const float* __restrict__ pos, __bf16* __restrict__ outLN, __bf16* __restrict__ outQK,
    int rows)
{
    int wave = threadIdx.x >> 5, lane = threadIdx.x & 31;
    int row = blockIdx.x * 8 + wave;
    if (row >= rows) return;
    const float* xr = x + (size_t)row * D_MODEL;
    float4 v0 = *(const float4*)(xr + lane * 8);
    float4 v1 = *(const float4*)(xr + lane * 8 + 4);
    float xs[8] = {v0.x, v0.y, v0.z, v0.w, v1.x, v1.y, v1.z, v1.w};
    float s = 0.f, s2 = 0.f;
    #pragma unroll
    for (int i = 0; i < 8; ++i) { s += xs[i]; s2 += xs[i] * xs[i]; }
    #pragma unroll
    for (int off = 16; off; off >>= 1) {
        s  += __shfl_xor(s,  off, 32);
        s2 += __shfl_xor(s2, off, 32);
    }
    float mu  = s * (1.f / 256.f);
    float var = s2 * (1.f / 256.f) - mu * mu;
    float inv = rsqrtf(var + 1e-5f);
    size_t base = (size_t)row * D_MODEL + lane * 8;
    #pragma unroll
    for (int i = 0; i < 8; ++i) {
        int c = lane * 8 + i;
        float t = (xs[i] - mu) * inv * g[c] + b[c];
        if (outLN) outLN[base + i] = (__bf16)t;
        if (outQK) outQK[base + i] = (__bf16)(t + pos[base + i]);
    }
}

// ---------------------------------------------------------------------------
// out_bf16 = a (+ b)   elementwise convert (4 per thread)
// ---------------------------------------------------------------------------
__global__ void add_cvt(const float* __restrict__ a, const float* __restrict__ b,
                        __bf16* __restrict__ out, int n) {
    int i = (blockIdx.x * blockDim.x + threadIdx.x) * 4;
    if (i >= n) return;
    float4 va = *(const float4*)(a + i);
    if (b) {
        float4 vb = *(const float4*)(b + i);
        va.x += vb.x; va.y += vb.y; va.z += vb.z; va.w += vb.w;
    }
    out[i]     = (__bf16)va.x;
    out[i + 1] = (__bf16)va.y;
    out[i + 2] = (__bf16)va.z;
    out[i + 3] = (__bf16)va.w;
}

__global__ void zero_f32(float* __restrict__ p, int n) {
    int i = blockIdx.x * blockDim.x + threadIdx.x;
    if (i < n) p[i] = 0.f;
}

// ---------------------------------------------------------------------------
// bf16 WMMA GEMM: C[M,N] = act(A[M,K] @ Wsw + bias) (+ resid)
// Block: 256 threads = 8 waves; each wave -> 16 rows x 64 cols (4 C-tiles).
// grid = (M/128, N/64).  act: 0=none, 1=elu+1, 2=relu
// A layout (contiguous, same k order as Wsw): lane<16 -> K {0..15}; lane>=16 -> K {16..31}
// ---------------------------------------------------------------------------
__global__ __launch_bounds__(256) void gemm_bf16_wmma(
    const __bf16* __restrict__ A, const __bf16* __restrict__ Wsw,
    const float* __restrict__ bias, const float* __restrict__ resid,
    __bf16* __restrict__ outB, float* __restrict__ outF,
    int M, int K, int N, int act)
{
    const int lane  = threadIdx.x & 31;
    const int wave  = threadIdx.x >> 5;
    const int mBase = blockIdx.x * 128 + wave * 16;
    const int nBase = blockIdx.y * 64;
    const int kcN   = K >> 5;

    v8f acc[4];
    #pragma unroll
    for (int t = 0; t < 4; ++t)
        #pragma unroll
        for (int r = 0; r < 8; ++r) acc[t][r] = 0.f;

    const int m = mBase + (lane & 15);
    const __bf16* aRow = A + (size_t)m * K + ((lane >> 4) ? 16 : 0);
    const int ntBase = nBase >> 4;

    for (int kc = 0; kc < kcN; ++kc) {
        bf16x8 lo = *(const bf16x8*)(aRow + kc * 32);
        bf16x8 hi = *(const bf16x8*)(aRow + kc * 32 + 8);
        v16bf a;
        #pragma unroll
        for (int i = 0; i < 8; ++i) { a[i] = lo[i]; a[i + 8] = hi[i]; }
        #pragma unroll
        for (int t = 0; t < 4; ++t) {
            const v16bf bmat = *(const v16bf*)(Wsw +
                ((((size_t)(ntBase + t) * kcN + kc) * 32 + lane) << 4));
            acc[t] = __builtin_amdgcn_wmma_f32_16x16x32_bf16(
                false, a, false, bmat, (short)0, acc[t], false, false);
        }
    }

    const int mrowAdd = (lane >> 4) << 3;
    #pragma unroll
    for (int t = 0; t < 4; ++t) {
        int n = nBase + t * 16 + (lane & 15);
        float bv = bias ? bias[n] : 0.f;
        #pragma unroll
        for (int r = 0; r < 8; ++r) {
            int mm = mBase + r + mrowAdd;
            float v = acc[t][r] + bv;
            if (act == 1) v = v > 0.f ? v + 1.f : __expf(v);   // elu(x)+1
            else if (act == 2) v = fmaxf(v, 0.f);              // relu
            size_t idx = (size_t)mm * N + n;
            if (resid) v += resid[idx];
            if (outF) outF[idx] = v;
            if (outB) outB[idx] = (__bf16)v;
        }
    }
}

// ---------------------------------------------------------------------------
// KV = sum_s K[s,:]^T (V[s,:]/L)  and  Ksum = sum_s K[s,:]  per (n,h).
// grid (8, 8, KV_SPLIT); LDS-staged 8 rows at a time; atomicAdd into fp32.
// ---------------------------------------------------------------------------
__global__ __launch_bounds__(256) void kv_accum(
    const __bf16* __restrict__ Kb, const __bf16* __restrict__ Vb,
    float* __restrict__ KV, float* __restrict__ Ksum)
{
    int n = blockIdx.x, h = blockIdx.y, sp = blockIdx.z;
    const int chunk = LSEQ / KV_SPLIT;
    int s0 = sp * chunk;
    __shared__ float KS[8][32];
    __shared__ float VS[8][32];
    int tid = threadIdx.x;
    float acc[4] = {0.f, 0.f, 0.f, 0.f};
    float kacc = 0.f;
    const float invS = 1.f / (float)LSEQ;

    for (int s = s0; s < s0 + chunk; s += 8) {
        __syncthreads();
        int ss = tid >> 5, c = tid & 31;
        size_t off = ((size_t)n * LSEQ + s + ss) * D_MODEL + h * HEAD_DIM + c;
        KS[ss][c] = (float)Kb[off];
        VS[ss][c] = (float)Vb[off] * invS;
        __syncthreads();
        #pragma unroll
        for (int q = 0; q < 8; ++q) {
            #pragma unroll
            for (int p = 0; p < 4; ++p) {
                int idx = tid + p * 256;
                acc[p] += KS[q][idx >> 5] * VS[q][idx & 31];
            }
            if (tid < 32) kacc += KS[q][tid];
        }
    }
    float* KVh = KV + (size_t)(n * NHEAD + h) * 1024;
    #pragma unroll
    for (int p = 0; p < 4; ++p) atomicAdd(&KVh[tid + p * 256], acc[p]);
    if (tid < 32) atomicAdd(&Ksum[(n * NHEAD + h) * 32 + tid], kacc);
}

// ---------------------------------------------------------------------------
// msg[l,h,v] = (Q[l,h,:] @ KV[h]) * L / (Q[l,h,:]·Ksum[h] + eps)
// One wave per (n,l,h) row; lane = output column v.
// ---------------------------------------------------------------------------
__global__ __launch_bounds__(256) void attn_out(
    const __bf16* __restrict__ Qb, const float* __restrict__ KV,
    const float* __restrict__ Ksum, __bf16* __restrict__ msg)
{
    int wave = threadIdx.x >> 5, lane = threadIdx.x & 31;
    int g  = blockIdx.x * 8 + wave;       // (n*L + l)*H + h
    int h  = g & (NHEAD - 1);
    int nl = g >> 3;
    int n  = nl / LSEQ;
    size_t qoff = (size_t)nl * D_MODEL + h * HEAD_DIM;
    float qv = (float)Qb[qoff + lane];
    int nh = n * NHEAD + h;
    float part = qv * Ksum[nh * 32 + lane];
    #pragma unroll
    for (int off = 16; off; off >>= 1) part += __shfl_xor(part, off, 32);
    float z = 1.f / (part + 1e-6f);
    const float* KVh = KV + (size_t)nh * 1024;
    float o = 0.f;
    #pragma unroll
    for (int d = 0; d < 32; ++d) {
        float qd = __shfl(qv, d, 32);
        o += qd * KVh[d * 32 + lane];
    }
    msg[qoff + lane] = (__bf16)(o * z * (float)LSEQ);
}

// ---------------------------------------------------------------------------
// Host orchestration
// ---------------------------------------------------------------------------
extern "C" void kernel_launch(void* const* d_in, const int* in_sizes, int n_in,
                              void* d_out, int out_size, void* d_ws, size_t ws_size,
                              hipStream_t stream) {
    const float* tgt     = (const float*)d_in[0];
    const float* memory  = (const float*)d_in[1];
    const float* tgt_pos = (const float*)d_in[2];
    const float* m_pos   = (const float*)d_in[3];
    const float* sa_w[4] = {(const float*)d_in[4], (const float*)d_in[6],
                            (const float*)d_in[8], (const float*)d_in[10]};
    const float* sa_b[3] = {(const float*)d_in[5], (const float*)d_in[7],
                            (const float*)d_in[9]};
    const float* ca_w[4] = {(const float*)d_in[11], (const float*)d_in[13],
                            (const float*)d_in[15], (const float*)d_in[17]};
    const float* ca_b[3] = {(const float*)d_in[12], (const float*)d_in[14],
                            (const float*)d_in[16]};
    const float* mlp_w1 = (const float*)d_in[18];
    const float* mlp_w2 = (const float*)d_in[19];
    const float* ln_g[3] = {(const float*)d_in[20], (const float*)d_in[22],
                            (const float*)d_in[24]};
    const float* ln_b[3] = {(const float*)d_in[21], (const float*)d_in[23],
                            (const float*)d_in[25]};

    const size_t EL = (size_t)MROWS * D_MODEL;   // 9,830,400 elements
    char* ws = (char*)d_ws;
    size_t off = 0;
    auto alloc = [&](size_t bytes) { size_t o = off; off += (bytes + 255) & ~(size_t)255; return o; };

    // swizzled weights (bf16): 8 x 256x256, mlp_w1 256x512, mlp_w2 512x256
    size_t wsw_off[10];
    for (int i = 0; i < 8; ++i) wsw_off[i] = alloc(65536 * 2);
    wsw_off[8] = alloc(131072 * 2);
    wsw_off[9] = alloc(131072 * 2);
    __bf16* wsw[10];
    for (int i = 0; i < 10; ++i) wsw[i] = (__bf16*)(ws + wsw_off[i]);

    float*  KVbuf  = (float*)(ws + alloc((65536 + 2048) * 4));  // KV + Ksum contiguous
    float*  KsumP  = KVbuf + 65536;
    __bf16* qkb    = (__bf16*)(ws + alloc(EL * 2));
    __bf16* t2b    = (__bf16*)(ws + alloc(EL * 2));             // also memory+m_pos
    __bf16* memb   = (__bf16*)(ws + alloc(EL * 2));
    __bf16* Qb     = (__bf16*)(ws + alloc(EL * 2));
    __bf16* Kb     = (__bf16*)(ws + alloc(EL * 2));
    __bf16* Vb     = (__bf16*)(ws + alloc(EL * 2));             // Kb..Vb contiguous -> hid alias
    __bf16* hid    = Kb;                                        // [MROWS,512] after attention dies
    __bf16* msgb   = (__bf16*)(ws + alloc(EL * 2));
    float*  tgtres = (float*)(ws + alloc(EL * 4));

    // --- weight prep ---
    const float* srcs[10] = {sa_w[0], sa_w[1], sa_w[2], sa_w[3],
                             ca_w[0], ca_w[1], ca_w[2], ca_w[3], mlp_w1, mlp_w2};
    const int wk[10] = {256,256,256,256,256,256,256,256,256,512};
    const int wn[10] = {256,256,256,256,256,256,256,256,512,256};
    for (int i = 0; i < 10; ++i) {
        int total = wk[i] * wn[i];
        prep_weight<<<(total + 255) / 256, 256, 0, stream>>>(srcs[i], wsw[i], wk[i], wn[i]);
    }

    dim3 g256(MROWS / 128, 4), g512(MROWS / 128, 8);

    // --- self-attention ---
    ln_fused<<<MROWS / 8, 256, 0, stream>>>(tgt, ln_g[0], ln_b[0], tgt_pos, t2b, qkb, MROWS);
    gemm_bf16_wmma<<<g256, 256, 0, stream>>>(qkb, wsw[0], sa_b[0], nullptr, Qb, nullptr, MROWS, 256, 256, 1);
    gemm_bf16_wmma<<<g256, 256, 0, stream>>>(qkb, wsw[1], sa_b[1], nullptr, Kb, nullptr, MROWS, 256, 256, 1);
    gemm_bf16_wmma<<<g256, 256, 0, stream>>>(t2b, wsw[2], sa_b[2], nullptr, Vb, nullptr, MROWS, 256, 256, 0);
    zero_f32<<<(67584 + 255) / 256, 256, 0, stream>>>(KVbuf, 67584);
    kv_accum<<<dim3(NBATCH, NHEAD, KV_SPLIT), 256, 0, stream>>>(Kb, Vb, KVbuf, KsumP);
    attn_out<<<MROWS * NHEAD / 8, 256, 0, stream>>>(Qb, KVbuf, KsumP, msgb);
    gemm_bf16_wmma<<<g256, 256, 0, stream>>>(msgb, wsw[3], nullptr, tgt, nullptr, tgtres, MROWS, 256, 256, 0);

    // --- cross-attention ---
    ln_fused<<<MROWS / 8, 256, 0, stream>>>(tgtres, ln_g[1], ln_b[1], tgt_pos, nullptr, qkb, MROWS);
    add_cvt<<<(int)(EL / 4 + 255) / 256, 256, 0, stream>>>(memory, m_pos, t2b, (int)EL);
    add_cvt<<<(int)(EL / 4 + 255) / 256, 256, 0, stream>>>(memory, nullptr, memb, (int)EL);
    gemm_bf16_wmma<<<g256, 256, 0, stream>>>(qkb, wsw[4], ca_b[0], nullptr, Qb, nullptr, MROWS, 256, 256, 1);
    gemm_bf16_wmma<<<g256, 256, 0, stream>>>(t2b, wsw[5], ca_b[1], nullptr, Kb, nullptr, MROWS, 256, 256, 1);
    gemm_bf16_wmma<<<g256, 256, 0, stream>>>(memb, wsw[6], ca_b[2], nullptr, Vb, nullptr, MROWS, 256, 256, 0);
    zero_f32<<<(67584 + 255) / 256, 256, 0, stream>>>(KVbuf, 67584);
    kv_accum<<<dim3(NBATCH, NHEAD, KV_SPLIT), 256, 0, stream>>>(Kb, Vb, KVbuf, KsumP);
    attn_out<<<MROWS * NHEAD / 8, 256, 0, stream>>>(Qb, KVbuf, KsumP, msgb);
    gemm_bf16_wmma<<<g256, 256, 0, stream>>>(msgb, wsw[7], nullptr, tgtres, nullptr, tgtres, MROWS, 256, 256, 0);

    // --- MLP ---
    ln_fused<<<MROWS / 8, 256, 0, stream>>>(tgtres, ln_g[2], ln_b[2], nullptr, t2b, nullptr, MROWS);
    gemm_bf16_wmma<<<g512, 256, 0, stream>>>(t2b, wsw[8], nullptr, nullptr, hid, nullptr, MROWS, 256, 512, 2);
    gemm_bf16_wmma<<<g256, 256, 0, stream>>>(hid, wsw[9], nullptr, tgtres, nullptr, (float*)d_out, MROWS, 512, 256, 0);
}
